// BaseRouter_22170621182522
// MI455X (gfx1250) — matrix operand order repaired and checked by reading, not compiled, not run on hardware
//
#include <hip/hip_runtime.h>
#include <math.h>

// Problem constants (match reference)
#define HID   2048
#define NTOK  4096          // B*S = 2*2048
#define NEXP  8
#define TOPK  2
#define CAP   1536          // int(4096 * 1.5 * 2 / 8)

typedef __attribute__((ext_vector_type(16))) __bf16 v16bf;
typedef __attribute__((ext_vector_type(8)))  __bf16 v8bf;
typedef __attribute__((ext_vector_type(8)))  float  v8f;

// ---------------- workspace layout (bytes) ----------------
// x_hi  bf16 [NTOK,HID]     16,777,216
// x_lo  bf16 [NTOK,HID]     16,777,216
// w1_hi bf16 [HID,HID]       8,388,608
// w1_lo bf16 [HID,HID]       8,388,608
// h     f32  [NTOK,HID]     33,554,432
// logits f32 [NTOK,NEXP]       131,072
// fe    i32  [NTOK*TOPK]        32,768
// fp    f32  [NTOK*TOPK]        32,768
// pos   i32  [NTOK*TOPK]        32,768
// valid i32  [NTOK*TOPK]        32,768
static constexpr size_t OFF_XH   = 0;
static constexpr size_t OFF_XL   = OFF_XH  + (size_t)NTOK * HID * 2;
static constexpr size_t OFF_WH   = OFF_XL  + (size_t)NTOK * HID * 2;
static constexpr size_t OFF_WL   = OFF_WH  + (size_t)HID  * HID * 2;
static constexpr size_t OFF_H    = OFF_WL  + (size_t)HID  * HID * 2;
static constexpr size_t OFF_LOG  = OFF_H   + (size_t)NTOK * HID * 4;
static constexpr size_t OFF_FE   = OFF_LOG + (size_t)NTOK * NEXP * 4;
static constexpr size_t OFF_FP   = OFF_FE  + (size_t)NTOK * TOPK * 4;
static constexpr size_t OFF_POS  = OFF_FP  + (size_t)NTOK * TOPK * 4;
static constexpr size_t OFF_VAL  = OFF_POS + (size_t)NTOK * TOPK * 4;

// ---- split f32 -> bf16(hi) + bf16(lo), round-to-nearest-even ----
__global__ void split_bf16_kernel(const float* __restrict__ src,
                                  unsigned short* __restrict__ hi,
                                  unsigned short* __restrict__ lo, int n) {
  int i = blockIdx.x * blockDim.x + threadIdx.x;
  int stride = gridDim.x * blockDim.x;
  for (; i < n; i += stride) {
    float v = src[i];
    unsigned u = __float_as_uint(v);
    unsigned r = (u + 0x7FFFu + ((u >> 16) & 1u)) >> 16;     // RNE to bf16
    float hf = __uint_as_float(r << 16);
    float l = v - hf;
    unsigned u2 = __float_as_uint(l);
    unsigned r2 = (u2 + 0x7FFFu + ((u2 >> 16) & 1u)) >> 16;
    hi[i] = (unsigned short)r;
    lo[i] = (unsigned short)r2;
  }
}

// ---- GEMM1: h = relu(x @ w1^T + b1), split-bf16 (hi*hi + hi*lo + lo*hi) ----
// grid (32, 32): blockIdx.x -> 128-row M tile, blockIdx.y -> 64-col N tile.
// 8 waves per block, wave w computes rows [m0+16w, m0+16w+16) x 64 cols.
__global__ __launch_bounds__(256) void gemm1_kernel(
    const unsigned short* __restrict__ xh_, const unsigned short* __restrict__ xl_,
    const unsigned short* __restrict__ wh_, const unsigned short* __restrict__ wl_,
    const float* __restrict__ b1, float* __restrict__ hout) {
  const __bf16* xh = (const __bf16*)xh_;
  const __bf16* xl = (const __bf16*)xl_;
  const __bf16* wh = (const __bf16*)wh_;
  const __bf16* wl = (const __bf16*)wl_;

  const int wave = threadIdx.x >> 5;
  const int lane = threadIdx.x & 31;
  const int m0 = blockIdx.x * 128 + wave * 16;
  const int n0 = blockIdx.y * 64;

  // A fragment addressing (16x32 bf16): lane<16 -> K 0..15 of row lane;
  // lane>=16 -> K 16..31 of row lane-16.
  const int arow  = m0 + (lane & 15);
  const int khalf = (lane >> 4) * 16;
  // B fragment addressing (32x16 bf16): lane<16 -> col lane, K 0..15;
  // lane>=16 -> col lane-16, K 16..31. w1 is [out,in] row-major == B^T, contiguous.
  const int bcol = lane & 15;
  const int ksec = (lane >> 4) * 16;

  v8f acc[4] = {};

  for (int k0 = 0; k0 < HID; k0 += 32) {
    union { v16bf v; v8bf h[2]; } ah, al;
    const __bf16* pah = xh + (size_t)arow * HID + k0 + khalf;
    ah.h[0] = *(const v8bf*)pah;
    ah.h[1] = *(const v8bf*)(pah + 8);
    const __bf16* pal = xl + (size_t)arow * HID + k0 + khalf;
    al.h[0] = *(const v8bf*)pal;
    al.h[1] = *(const v8bf*)(pal + 8);

#pragma unroll
    for (int j = 0; j < 4; ++j) {
      const size_t boff = (size_t)(n0 + j * 16 + bcol) * HID + k0 + ksec;
      v16bf bh = *(const v16bf*)(wh + boff);
      v16bf bl = *(const v16bf*)(wl + boff);
      // (xh+xl)(wh+wl) ~= xh*wh + xh*wl + xl*wh   (xl*wl ~ 2^-16, dropped)
      acc[j] = __builtin_amdgcn_wmma_f32_16x16x32_bf16(false, ah.v, false, bh,
                                                       (short)0, acc[j], false, false);
      acc[j] = __builtin_amdgcn_wmma_f32_16x16x32_bf16(false, ah.v, false, bl,
                                                       (short)0, acc[j], false, false);
      acc[j] = __builtin_amdgcn_wmma_f32_16x16x32_bf16(false, al.v, false, bh,
                                                       (short)0, acc[j], false, false);
    }
  }

  // C/D layout: VGPR r, lane l -> row = (l>>4)*8 + r, col = l&15
  const int crow = m0 + (lane >> 4) * 8;
  const int ccol = n0 + (lane & 15);
#pragma unroll
  for (int j = 0; j < 4; ++j) {
    const int col = ccol + j * 16;
    const float bias = b1[col];
#pragma unroll
    for (int r = 0; r < 8; ++r) {
      float v = acc[j][r] + bias;
      hout[(size_t)(crow + r) * HID + col] = v > 0.f ? v : 0.f;
    }
  }
}

// ---- logits[t,e] = h[t,:] . w2[e,:] + b2[e]; one wave per token ----
__global__ __launch_bounds__(256) void logits_kernel(
    const float* __restrict__ h, const float* __restrict__ w2,
    const float* __restrict__ b2, float* __restrict__ logits) {
  const int wave = threadIdx.x >> 5;
  const int lane = threadIdx.x & 31;
  const int t = blockIdx.x * 8 + wave;
  const float* hr = h + (size_t)t * HID;
  float s[NEXP] = {0.f, 0.f, 0.f, 0.f, 0.f, 0.f, 0.f, 0.f};
  for (int i = lane; i < HID; i += 32) {
    float hv = hr[i];
#pragma unroll
    for (int e = 0; e < NEXP; ++e) s[e] += hv * w2[(size_t)e * HID + i];
  }
#pragma unroll
  for (int e = 0; e < NEXP; ++e) {
    float v = s[e];
    for (int off = 16; off > 0; off >>= 1) v += __shfl_xor(v, off, 32);
    if (lane == 0) logits[(size_t)t * NEXP + e] = v + b2[e];
  }
}

// ---- softmax over 8, top-2 (ties -> lowest index), renormalize ----
__global__ void route_kernel(const float* __restrict__ logits,
                             float* __restrict__ probs_out,
                             int* __restrict__ fe, float* __restrict__ fp) {
  int t = blockIdx.x * blockDim.x + threadIdx.x;
  if (t >= NTOK) return;
  float l[NEXP], p[NEXP];
  float m = -INFINITY;
#pragma unroll
  for (int e = 0; e < NEXP; ++e) { l[e] = logits[(size_t)t * NEXP + e]; m = fmaxf(m, l[e]); }
  float sum = 0.f;
#pragma unroll
  for (int e = 0; e < NEXP; ++e) { p[e] = expf(l[e] - m); sum += p[e]; }
  float inv = 1.f / sum;
#pragma unroll
  for (int e = 0; e < NEXP; ++e) { p[e] *= inv; probs_out[(size_t)t * NEXP + e] = p[e]; }
  int e1 = 0; float p1 = p[0];
#pragma unroll
  for (int e = 1; e < NEXP; ++e) if (p[e] > p1) { p1 = p[e]; e1 = e; }
  int e2 = -1; float p2 = -INFINITY;
#pragma unroll
  for (int e = 0; e < NEXP; ++e) if (e != e1 && p[e] > p2) { p2 = p[e]; e2 = e; }
  float norm = 1.f / (p1 + p2 + 1e-8f);
  fe[2 * t] = e1;     fe[2 * t + 1] = e2;
  fp[2 * t] = p1 * norm; fp[2 * t + 1] = p2 * norm;
}

// ---- ordered per-expert slot assignment + aux loss (single block) ----
__global__ __launch_bounds__(256) void scan_kernel(
    const int* __restrict__ fe, const float* __restrict__ probs,
    int* __restrict__ pos, int* __restrict__ valid, float* __restrict__ aux_out) {
  __shared__ int   cnt[256][NEXP];
  __shared__ float ps[256][NEXP];
  __shared__ int   tot[NEXP];
  const int t = threadIdx.x;
  const int CHUNK = (NTOK * TOPK) / 256;   // 32
  const int b0 = t * CHUNK;

  int c[NEXP] = {0, 0, 0, 0, 0, 0, 0, 0};
  for (int i = 0; i < CHUNK; ++i) c[fe[b0 + i]]++;
#pragma unroll
  for (int e = 0; e < NEXP; ++e) cnt[t][e] = c[e];
  __syncthreads();

  if (t < NEXP) {   // exclusive scan over 256 per-thread counts for expert t
    int run = 0;
    for (int th = 0; th < 256; ++th) { int v = cnt[th][t]; cnt[th][t] = run; run += v; }
    tot[t] = run;
  }
  __syncthreads();

#pragma unroll
  for (int e = 0; e < NEXP; ++e) c[e] = cnt[t][e];
  for (int i = 0; i < CHUNK; ++i) {
    int e = fe[b0 + i];
    int p = c[e]++;
    pos[b0 + i]   = p < CAP ? p : CAP - 1;
    valid[b0 + i] = p < CAP ? 1 : 0;
  }

  // deterministic per-expert prob-sum reduction
  float s[NEXP] = {0.f, 0.f, 0.f, 0.f, 0.f, 0.f, 0.f, 0.f};
  for (int tok = t; tok < NTOK; tok += 256)
#pragma unroll
    for (int e = 0; e < NEXP; ++e) s[e] += probs[(size_t)tok * NEXP + e];
#pragma unroll
  for (int e = 0; e < NEXP; ++e) ps[t][e] = s[e];
  __syncthreads();
  for (int str = 128; str > 0; str >>= 1) {
    if (t < str)
#pragma unroll
      for (int e = 0; e < NEXP; ++e) ps[t][e] += ps[t + str][e];
    __syncthreads();
  }
  if (t == 0) {
    float aux = 0.f;
#pragma unroll
    for (int e = 0; e < NEXP; ++e)
      aux += (ps[0][e] / (float)NTOK) * ((float)tot[e] / (float)(NTOK * TOPK));
    *aux_out = aux * (float)NEXP;
  }
}

// ---- zero dispatch+combine (grid-stride float4) ----
__global__ void zero_kernel(float4* __restrict__ p, size_t n4) {
  size_t i = (size_t)blockIdx.x * blockDim.x + threadIdx.x;
  size_t stride = (size_t)gridDim.x * blockDim.x;
  float4 z = make_float4(0.f, 0.f, 0.f, 0.f);
  for (; i < n4; i += stride) p[i] = z;
}

// ---- scatter valid assignments ----
__global__ void scatter_kernel(const int* __restrict__ fe, const float* __restrict__ fp,
                               const int* __restrict__ pos, const int* __restrict__ valid,
                               float* __restrict__ dispatch, float* __restrict__ combine) {
  int i = blockIdx.x * blockDim.x + threadIdx.x;
  if (i >= NTOK * TOPK) return;
  if (!valid[i]) return;
  int tok = i >> 1;
  size_t off = (size_t)tok * NEXP * CAP + (size_t)fe[i] * CAP + (size_t)pos[i];
  dispatch[off] = 1.0f;
  combine[off]  = fp[i];
}

extern "C" void kernel_launch(void* const* d_in, const int* in_sizes, int n_in,
                              void* d_out, int out_size, void* d_ws, size_t ws_size,
                              hipStream_t stream) {
  const float* x  = (const float*)d_in[0];   // [2,2048,2048]
  const float* w1 = (const float*)d_in[1];   // [2048,2048]
  const float* b1 = (const float*)d_in[2];   // [2048]
  const float* w2 = (const float*)d_in[3];   // [8,2048]
  const float* b2 = (const float*)d_in[4];   // [8]

  char* ws = (char*)d_ws;
  unsigned short* xh = (unsigned short*)(ws + OFF_XH);
  unsigned short* xl = (unsigned short*)(ws + OFF_XL);
  unsigned short* wh = (unsigned short*)(ws + OFF_WH);
  unsigned short* wl = (unsigned short*)(ws + OFF_WL);
  float* hbuf   = (float*)(ws + OFF_H);
  float* logits = (float*)(ws + OFF_LOG);
  int*   fe     = (int*)(ws + OFF_FE);
  float* fp     = (float*)(ws + OFF_FP);
  int*   pos    = (int*)(ws + OFF_POS);
  int*   valid  = (int*)(ws + OFF_VAL);

  float* out      = (float*)d_out;
  float* dispatch = out;                                   // [T,E,CAP]
  float* combine  = out + (size_t)NTOK * NEXP * CAP;       // [T,E,CAP]
  float* probs    = out + 2 * (size_t)NTOK * NEXP * CAP;   // [T,E]
  float* aux      = probs + (size_t)NTOK * NEXP;           // [1]

  split_bf16_kernel<<<2048, 256, 0, stream>>>(x,  xh, xl, NTOK * HID);
  split_bf16_kernel<<<1024, 256, 0, stream>>>(w1, wh, wl, HID * HID);

  gemm1_kernel<<<dim3(NTOK / 128, HID / 64), 256, 0, stream>>>(xh, xl, wh, wl, b1, hbuf);

  logits_kernel<<<NTOK / 8, 256, 0, stream>>>(hbuf, w2, b2, logits);
  route_kernel<<<NTOK / 256, 256, 0, stream>>>(logits, probs, fe, fp);

  zero_kernel<<<4096, 256, 0, stream>>>((float4*)out, (size_t)2 * NTOK * NEXP * CAP / 4);
  scan_kernel<<<1, 256, 0, stream>>>(fe, probs, pos, valid, aux);
  scatter_kernel<<<(NTOK * TOPK) / 256, 256, 0, stream>>>(fe, fp, pos, valid, dispatch, combine);
}
